// MainGCN_61340722921801
// MI455X (gfx1250) — matrix operand order repaired and checked by reading, not compiled, not run on hardware
//
#include <hip/hip_runtime.h>

// ---------------- problem constants (from reference) ----------------
#define G_GRAPHS 64
#define N_NODES  1024
#define E_EDGES  16384   // per graph; == 1<<14

typedef __attribute__((ext_vector_type(16))) __bf16 v16bf;
typedef __attribute__((ext_vector_type(8)))  float  v8f;

union Frag16 { unsigned int u[8]; v16bf v; };

#define LDS_STRIDE 20   // dwords; 80B row = 16B aligned; 20n mod 64 distinct for n=0..15

// round-to-nearest-even fp32 -> bf16, packed pair
__device__ __forceinline__ unsigned int pack2_bf16(float lo, float hi) {
    unsigned int a = __float_as_uint(lo);
    unsigned int b = __float_as_uint(hi);
    a = (a + 0x7FFFu + ((a >> 16) & 1u)) >> 16;
    b = (b + 0x7FFFu + ((b >> 16) & 1u)) >> 16;
    return (a & 0xFFFFu) | (b << 16);
}

// ---------------- zero fill ----------------
__global__ void gcn_zero_f32(float* __restrict__ p, long n) {
    long i = blockIdx.x * (long)blockDim.x + threadIdx.x;
    if (i < n) p[i] = 0.0f;
}

// ---------------- weight pre-convert: W[K][N] f32 -> Wt[N][K/2] packed bf16 ----
__global__ void gcn_convert_wt(const float* __restrict__ W,
                               unsigned* __restrict__ Wt, int K, int N) {
    long i = blockIdx.x * (long)blockDim.x + threadIdx.x;
    long total = (long)N * (K >> 1);
    if (i >= total) return;
    int n  = (int)(i / (K >> 1));
    int kp = (int)(i - (long)n * (K >> 1));
    float lo = W[(size_t)(2 * kp)     * N + n];
    float hi = W[(size_t)(2 * kp + 1) * N + n];
    Wt[i] = pack2_bf16(lo, hi);
}

// ---------------- edge scatter-add (segment_sum) ----------------
__global__ void gcn_scatter_add(const float* __restrict__ h,
                                const int*   __restrict__ eidx,
                                float*       __restrict__ agg,
                                int F) {
    long tid = blockIdx.x * (long)blockDim.x + threadIdx.x;
    int  perEdge = F >> 2;                       // float4 per thread
    long eg = tid / perEdge;                     // global edge id, < G*E
    int  f  = (int)(tid - eg * perEdge) * 4;
    int  g  = (int)(eg >> 14);                   // E = 1<<14
    int  e  = (int)(eg & (E_EDGES - 1));
    const int* base = eidx + (long)g * 2 * E_EDGES;
    int s = base[e];
    int d = base[E_EDGES + e];
    const float4 v = *(const float4*)(h + ((long)g * N_NODES + s) * F + f);
    float* dp = agg + ((long)g * N_NODES + d) * F + f;
    unsafeAtomicAdd(dp + 0, v.x);
    unsafeAtomicAdd(dp + 1, v.y);
    unsafeAtomicAdd(dp + 2, v.z);
    unsafeAtomicAdd(dp + 3, v.w);
}

// ---------------- dual-operand bf16-WMMA GEMM, double-buffered + async W ----
// out = epilogue( A1@W1 [+ A2@W2] + bias )
// A*: [M,K] f32 row-major.  Wt*: [Ntot][K/2] packed-bf16 dwords (n-major W^T).
__global__ __launch_bounds__(256)
void gcn_gemm_wmma(const float* __restrict__ A1, const unsigned* __restrict__ Wt1,
                   const float* __restrict__ A2, const unsigned* __restrict__ Wt2,
                   const float* __restrict__ bias, float* __restrict__ out,
                   int M, int Ntot, int K, int act, int pool, float poolScale) {
    __shared__ unsigned sA[2][128 * LDS_STRIDE];   // [row][kpair]
    __shared__ unsigned sW[2][128 * LDS_STRIDE];   // [n][kpair]

    const int tid  = threadIdx.x;
    const int lane = tid & 31;
    const int l15  = lane & 15;
    const int lhi  = (lane >> 4) & 1;
    const int wid  = tid >> 5;              // 8 waves
    const int wMb  = (wid & 1) * 64;
    const int wNb  = (wid >> 1) * 32;
    const int mblock = blockIdx.x * 128;
    const int nblock = blockIdx.y * 128;
    const int spp    = K >> 5;              // K-steps per pass
    const int npass  = (A2 != nullptr) ? 2 : 1;
    const int total  = npass * spp;
    const int Khalf  = K >> 1;
    const bool fullM = (mblock + 128 <= M); // uniform: skip per-load row guards

    v8f zero = {};
    v8f acc[4][2];
#pragma unroll
    for (int t = 0; t < 4; ++t)
#pragma unroll
        for (int u = 0; u < 2; ++u) acc[t][u] = zero;

    auto issueTile = [&](int step, int buf) {
        const float*    A  = (step >= spp) ? A2  : A1;
        const unsigned* Wt = (step >= spp) ? Wt2 : Wt1;
        int kb = ((step >= spp) ? (step - spp) : step) << 5;
        // A tile: 128 rows x 16 bf16-pairs (load f32x2, convert, ds_store)
        if (fullM) {
#pragma unroll
            for (int i = 0; i < 8; ++i) {
                int idx = tid + i * 256;
                int row = idx >> 4;
                int p   = idx & 15;
                float2 val = *(const float2*)(A + (size_t)(mblock + row) * K + kb + p * 2);
                sA[buf][row * LDS_STRIDE + p] = pack2_bf16(val.x, val.y);
            }
        } else {
#pragma unroll
            for (int i = 0; i < 8; ++i) {
                int idx = tid + i * 256;
                int row = idx >> 4;
                int p   = idx & 15;
                int mrow = mblock + row;
                float2 val = make_float2(0.f, 0.f);
                if (mrow < M)
                    val = *(const float2*)(A + (size_t)mrow * K + kb + p * 2);
                sA[buf][row * LDS_STRIDE + p] = pack2_bf16(val.x, val.y);
            }
        }
        // W tile: byte-exact copy of pre-packed bf16 -> CDNA5 async load to LDS.
        // 128 rows x 64B, as 512 b128 transfers (2 per thread).
#pragma unroll
        for (int i = 0; i < 2; ++i) {
            int idx = tid + i * 256;
            int n   = idx >> 2;             // 0..127
            int seg = idx & 3;              // 16B segment within row
            unsigned ldsAddr = (unsigned)(size_t)(&sW[buf][n * LDS_STRIDE + seg * 4]);
            unsigned long long gAddr = (unsigned long long)(size_t)
                (Wt + (size_t)(nblock + n) * Khalf + (kb >> 1) + seg * 4);
            asm volatile("global_load_async_to_lds_b128 %0, %1, off"
                         :: "v"(ldsAddr), "v"(gAddr) : "memory");
        }
    };

    issueTile(0, 0);

    for (int s = 0; s < total; ++s) {
        int buf = s & 1;
        asm volatile("s_wait_asynccnt 0" ::: "memory");  // own async W writes done
        __syncthreads();                                 // all waves' tile data visible
        if (s + 1 < total) issueTile(s + 1, buf ^ 1);

        // fragments per ISA 7.12.2 (16-bit A 16x32; B mirrored striping)
        Frag16 fa[4], fb[2];
#pragma unroll
        for (int t = 0; t < 4; ++t)
#pragma unroll
            for (int v = 0; v < 8; ++v)
                fa[t].u[v] = sA[buf][(wMb + t * 16 + l15) * LDS_STRIDE +
                                     ((v >> 2) * 8 + lhi * 4 + (v & 3))];
#pragma unroll
        for (int u = 0; u < 2; ++u)
#pragma unroll
            for (int v = 0; v < 8; ++v)
                fb[u].u[v] = sW[buf][(wNb + u * 16 + l15) * LDS_STRIDE +
                                     ((v >> 2) * 8 + lhi * 4 + (v & 3))];
#pragma unroll
        for (int t = 0; t < 4; ++t)
#pragma unroll
            for (int u = 0; u < 2; ++u)
                acc[t][u] = __builtin_amdgcn_wmma_f32_16x16x32_bf16(
                    false, fa[t].v, false, fb[u].v,
                    (short)0, acc[t][u], false, false);
    }

    // epilogue: C/D layout -> VGPR r holds M = r + 8*lhi, N = l15
#pragma unroll
    for (int t = 0; t < 4; ++t) {
#pragma unroll
        for (int r = 0; r < 8; ++r) {
            int row = mblock + wMb + t * 16 + lhi * 8 + r;
            if (row < M) {
#pragma unroll
                for (int u = 0; u < 2; ++u) {
                    int col = nblock + wNb + u * 16 + l15;
                    float v = acc[t][u][r] + bias[col];
                    if (act) v = fmaxf(v, 0.0f);
                    if (pool)
                        unsafeAtomicAdd(out + (size_t)(row >> 10) * Ntot + col,
                                        v * poolScale);
                    else
                        out[(size_t)row * Ntot + col] = v;
                }
            }
        }
    }
}

// ---------------- host orchestration ----------------
extern "C" void kernel_launch(void* const* d_in, const int* in_sizes, int n_in,
                              void* d_out, int out_size, void* d_ws, size_t ws_size,
                              hipStream_t stream) {
    const float* x       = (const float*)d_in[0];
    const int*   eidx    = (const int*)  d_in[1];
    const float* W1_rel  = (const float*)d_in[2];
    const float* W1_root = (const float*)d_in[3];
    const float* b1      = (const float*)d_in[4];
    const float* W2_rel  = (const float*)d_in[5];
    const float* W2_root = (const float*)d_in[6];
    const float* b2      = (const float*)d_in[7];
    const float* fc_W    = (const float*)d_in[8];
    const float* fc_b    = (const float*)d_in[9];
    float* out = (float*)d_out;

    const int M   = G_GRAPHS * N_NODES;   // 65536
    const int IND = 512, HID = 1024, OUTD = 1024, FCD = 2048;

    // workspace: agg | h1 | pooled | packed-bf16 transposed weights
    float* agg    = (float*)d_ws;
    float* h1     = agg + (size_t)M * HID;
    float* pooled = h1  + (size_t)M * HID;
    unsigned* wt1r = (unsigned*)(pooled + (size_t)G_GRAPHS * OUTD);
    unsigned* wt1o = wt1r + (size_t)HID  * (IND  >> 1);   // [1024][256]
    unsigned* wt2r = wt1o + (size_t)HID  * (IND  >> 1);
    unsigned* wt2o = wt2r + (size_t)OUTD * (HID  >> 1);   // [1024][512]
    unsigned* fcwt = wt2o + (size_t)OUTD * (HID  >> 1);   // [2048][512]

    // ---- weight pre-conversion (once per launch; tiny) ----
    long c1 = (long)HID  * (IND >> 1);
    long c2 = (long)OUTD * (HID >> 1);
    long c3 = (long)FCD  * (OUTD >> 1);
    gcn_convert_wt<<<(int)((c1 + 255) / 256), 256, 0, stream>>>(W1_rel,  wt1r, IND,  HID);
    gcn_convert_wt<<<(int)((c1 + 255) / 256), 256, 0, stream>>>(W1_root, wt1o, IND,  HID);
    gcn_convert_wt<<<(int)((c2 + 255) / 256), 256, 0, stream>>>(W2_rel,  wt2r, HID,  OUTD);
    gcn_convert_wt<<<(int)((c2 + 255) / 256), 256, 0, stream>>>(W2_root, wt2o, HID,  OUTD);
    gcn_convert_wt<<<(int)((c3 + 255) / 256), 256, 0, stream>>>(fc_W,    fcwt, OUTD, FCD);

    // ---- layer 1 ----
    long nAgg1 = (long)M * IND;
    gcn_zero_f32<<<(int)((nAgg1 + 255) / 256), 256, 0, stream>>>(agg, nAgg1);
    long tSc1 = (long)G_GRAPHS * E_EDGES * (IND / 4);
    gcn_scatter_add<<<(int)(tSc1 / 256), 256, 0, stream>>>(x, eidx, agg, IND);
    dim3 g1((M + 127) / 128, HID / 128);
    gcn_gemm_wmma<<<g1, 256, 0, stream>>>(agg, wt1r, x, wt1o, b1, h1,
                                          M, HID, IND, /*act=*/1, /*pool=*/0, 0.f);

    // ---- layer 2 (fused mean-pool epilogue) ----
    long nAgg2 = (long)M * HID;
    gcn_zero_f32<<<(int)((nAgg2 + 255) / 256), 256, 0, stream>>>(agg, nAgg2);
    long tSc2 = (long)G_GRAPHS * E_EDGES * (HID / 4);
    gcn_scatter_add<<<(int)(tSc2 / 256), 256, 0, stream>>>(h1, eidx, agg, HID);
    long nPool = (long)G_GRAPHS * OUTD;
    gcn_zero_f32<<<(int)((nPool + 255) / 256), 256, 0, stream>>>(pooled, nPool);
    dim3 g2((M + 127) / 128, OUTD / 128);
    gcn_gemm_wmma<<<g2, 256, 0, stream>>>(agg, wt2r, h1, wt2o, b2, pooled,
                                          M, OUTD, HID, /*act=*/0, /*pool=*/1,
                                          1.0f / (float)N_NODES);

    // ---- final FC: [64,1024] @ [1024,2048] + b -> d_out ----
    dim3 g3(1, FCD / 128);
    gcn_gemm_wmma<<<g3, 256, 0, stream>>>(pooled, fcwt, nullptr, nullptr, fc_b, out,
                                          G_GRAPHS, FCD, OUTD, /*act=*/0, /*pool=*/0, 0.f);
}
